// SelfAttention_38577396253319
// MI455X (gfx1250) — matrix operand order repaired and checked by reading, not compiled, not run on hardware
//
#include <hip/hip_runtime.h>
#include <hip/hip_bf16.h>

#define N_NODES  50000
#define N_EDGES  800000
#define N_GRAPHS 256
#define F_IN     256
#define H_DIM    64

typedef __attribute__((ext_vector_type(16))) _Float16 v16h;
typedef __attribute__((ext_vector_type(4)))  _Float16 v4h;
typedef __attribute__((ext_vector_type(8)))  float    v8f;

// ---------------------------------------------------------------------------
// Kernel 0a: zero out-accumulator + per-graph denominators (scatter targets)
// ---------------------------------------------------------------------------
__global__ __launch_bounds__(256)
void zero_kernel(float* __restrict__ out, float* __restrict__ denom) {
    int idx = blockIdx.x * 256 + threadIdx.x;
    if (idx < N_NODES * H_DIM) out[idx] = 0.0f;
    if (idx < N_GRAPHS)        denom[idx] = 0.0f;
}

// ---------------------------------------------------------------------------
// Kernel 0b: transpose + convert weights to f16:  Wt[m][n][k] = (f16) W_m[k][n]
// Makes the WMMA B-fragment reads contiguous (two b128 loads per lane).
// ---------------------------------------------------------------------------
__global__ __launch_bounds__(256)
void wt_kernel(const float* __restrict__ Wq, const float* __restrict__ Wk,
               const float* __restrict__ Wv, _Float16* __restrict__ Wt) {
    int idx = blockIdx.x * 256 + threadIdx.x;            // over 3*64*256
    if (idx >= 3 * H_DIM * F_IN) return;
    int m   = idx / (H_DIM * F_IN);
    int rem = idx % (H_DIM * F_IN);
    int n   = rem / F_IN;
    int kk  = rem % F_IN;
    const float* W = (m == 0) ? Wq : (m == 1) ? Wk : Wv;
    Wt[idx] = (_Float16)W[(size_t)kk * H_DIM + n];
}

// ---------------------------------------------------------------------------
// Kernel 0c: convert x to f16 once (A side of the GEMM).  Removes all
// f32->f16 cvts from the WMMA hot loop and cuts A-side bytes in half.
// ---------------------------------------------------------------------------
__global__ __launch_bounds__(256)
void xh_kernel(const float* __restrict__ x, _Float16* __restrict__ xh) {
    int idx = blockIdx.x * 256 + threadIdx.x;            // over N*F/4
    if (idx >= (N_NODES * F_IN) / 4) return;
    const float4 f = *(const float4*)(x + (size_t)idx * 4);
    v4h h;
    h[0] = (_Float16)f.x; h[1] = (_Float16)f.y;
    h[2] = (_Float16)f.z; h[3] = (_Float16)f.w;
    *(v4h*)(xh + (size_t)idx * 4) = h;
}

// ---------------------------------------------------------------------------
// Kernel 1: fused q/k/v projection GEMM with v_wmma_f32_16x16x32_f16.
// One wave per 16-node row tile; all THREE matrices computed in the same
// wave so each A fragment is reused by 12 WMMAs (4 N-tiles x 3 matrices).
// C operand preloaded with bias -> D = x@W + b in one pass.
// 12 x 8 = 96 accumulator VGPRs; 96 wmma per wave.
// ---------------------------------------------------------------------------
__global__ __launch_bounds__(32)
void proj_gemm_kernel(const _Float16* __restrict__ xh,   // [N][256] f16
                      const _Float16* __restrict__ Wt,   // [3][64][256] f16, W^T
                      const float* __restrict__ bq,
                      const float* __restrict__ bk,
                      const float* __restrict__ bv,
                      float* __restrict__ q,
                      float* __restrict__ k,
                      float* __restrict__ v) {
    const int rowTile = blockIdx.x;         // 0..3124
    const int lane    = threadIdx.x;        // wave32
    const int hi      = lane >> 4;          // lane half
    const int l16     = lane & 15;
    const int rowBase = rowTile * 16;

    const float* biases[3] = { bq, bk, bv };
    float*       outs[3]   = { q,  k,  v  };

    // Accumulators: [matrix][N-tile]; every element of an 8-VGPR C tile for
    // this lane has column N = nt*16 + l16 -> initialize with bias[N].
    v8f acc[3][4];
#pragma unroll
    for (int m = 0; m < 3; ++m) {
#pragma unroll
        for (int nt = 0; nt < 4; ++nt) {
            float b = biases[m][nt * 16 + l16];
#pragma unroll
            for (int r = 0; r < 8; ++r) acc[m][nt][r] = b;
        }
    }

    // A rows: this lane owns node row M = l16 of the tile.
    const _Float16* xrow = xh + (size_t)(rowBase + l16) * F_IN;

#pragma unroll
    for (int ks = 0; ks < 8; ++ks) {        // K = 256 in steps of 32
        const int kb = ks * 32;

        // A fragment (16-bit A 16x32 layout): lane half 'hi' holds
        // K = kb + hi*8 + {0..7} in elems 0..7 and + {16..23} in elems 8..15.
        // f16 source -> two contiguous b128 loads, no cvts.
        v16h a;
        const _Float16* xp = xrow + kb + hi * 8;
#pragma unroll
        for (int i = 0; i < 8; ++i) {
            a[i]     = xp[i];
            a[8 + i] = xp[16 + i];
        }

#pragma unroll
        for (int m = 0; m < 3; ++m) {
            const _Float16* W = Wt + (size_t)m * H_DIM * F_IN;
#pragma unroll
            for (int nt = 0; nt < 4; ++nt) {
                // B fragment (16-bit B KxN layout): lane column N = nt*16+l16,
                // lane half hi holds K = kb + hi*8 + {0..7} and + {16..23}.
                // W^T[n][k] contiguous in k -> 2x b128 loads.
                const _Float16* wp =
                    W + (size_t)(nt * 16 + l16) * F_IN + kb + hi * 8;
                v16h b;
#pragma unroll
                for (int i = 0; i < 8; ++i) {
                    b[i]     = wp[i];
                    b[8 + i] = wp[16 + i];
                }

                acc[m][nt] = __builtin_amdgcn_wmma_f32_16x16x32_f16(
                    false, a, false, b, (short)0, acc[m][nt], false, false);
            }
        }
    }

    // Store D: VGPR r -> node row (r + hi*8), column nt*16 + l16.
#pragma unroll
    for (int m = 0; m < 3; ++m) {
        float* outp = outs[m];
#pragma unroll
        for (int nt = 0; nt < 4; ++nt) {
            const int col = nt * 16 + l16;
#pragma unroll
            for (int r = 0; r < 8; ++r) {
                const int node = rowBase + hi * 8 + r;
                outp[(size_t)node * H_DIM + col] = acc[m][nt][r];
            }
        }
    }
}

// ---------------------------------------------------------------------------
// Kernel 2: per-edge attention score + per-graph exp segment-sum.
// 16 lanes per edge (float4 gather per lane), shfl-xor reduction within 16.
// q/k (25.6 MB) are L2-resident on MI455X (192 MB L2) -> L2-bound gathers.
// ---------------------------------------------------------------------------
__global__ __launch_bounds__(256)
void score_kernel(const float* __restrict__ q, const float* __restrict__ k,
                  const int* __restrict__ src, const int* __restrict__ dst,
                  const int* __restrict__ batch,
                  float* __restrict__ expbuf, float* __restrict__ denom) {
    int t = blockIdx.x * 256 + threadIdx.x;
    int e = t >> 4;
    int l = t & 15;
    if (e >= N_EDGES) return;
    int s = src[e];
    int d = dst[e];

    const float4 qv = *(const float4*)(q + (size_t)d * H_DIM + l * 4);
    const float4 kv = *(const float4*)(k + (size_t)s * H_DIM + l * 4);
    float dp = qv.x * kv.x + qv.y * kv.y + qv.z * kv.z + qv.w * kv.w;
#pragma unroll
    for (int m = 8; m >= 1; m >>= 1) dp += __shfl_xor(dp, m, 32);

    if (l == 0) {
        float ex = __expf(dp * 0.125f);      // 1/sqrt(64)
        expbuf[e] = ex;
        __hip_atomic_fetch_add(&denom[batch[s]], ex,
                               __ATOMIC_RELAXED, __HIP_MEMORY_SCOPE_AGENT);
    }
}

// ---------------------------------------------------------------------------
// Kernel 3: out[dest] += v[src] * a  (scatter-add, native f32 atomics in L2)
// ---------------------------------------------------------------------------
__global__ __launch_bounds__(256)
void agg_kernel(const float* __restrict__ v,
                const int* __restrict__ src, const int* __restrict__ dst,
                const int* __restrict__ batch,
                const float* __restrict__ expbuf, const float* __restrict__ denom,
                float* __restrict__ out) {
    int t = blockIdx.x * 256 + threadIdx.x;
    int e = t >> 4;
    int l = t & 15;
    if (e >= N_EDGES) return;
    int s = src[e];
    int d = dst[e];

    float a = expbuf[e] / (denom[batch[s]] + 1e-6f);
    const float4 vv = *(const float4*)(v + (size_t)s * H_DIM + l * 4);
    float* op = out + (size_t)d * H_DIM + l * 4;
    __hip_atomic_fetch_add(op + 0, vv.x * a, __ATOMIC_RELAXED, __HIP_MEMORY_SCOPE_AGENT);
    __hip_atomic_fetch_add(op + 1, vv.y * a, __ATOMIC_RELAXED, __HIP_MEMORY_SCOPE_AGENT);
    __hip_atomic_fetch_add(op + 2, vv.z * a, __ATOMIC_RELAXED, __HIP_MEMORY_SCOPE_AGENT);
    __hip_atomic_fetch_add(op + 3, vv.w * a, __ATOMIC_RELAXED, __HIP_MEMORY_SCOPE_AGENT);
}

// ---------------------------------------------------------------------------
extern "C" void kernel_launch(void* const* d_in, const int* in_sizes, int n_in,
                              void* d_out, int out_size, void* d_ws, size_t ws_size,
                              hipStream_t stream) {
    // setup_inputs order: x, Wq, bq, Wk, bk, Wv, bv, edge_index, batch
    const float* x   = (const float*)d_in[0];
    const float* Wq  = (const float*)d_in[1];
    const float* bq  = (const float*)d_in[2];
    const float* Wk  = (const float*)d_in[3];
    const float* bk  = (const float*)d_in[4];
    const float* Wv  = (const float*)d_in[5];
    const float* bv  = (const float*)d_in[6];
    const int*   ei  = (const int*)d_in[7];          // [2, E]
    const int*   bat = (const int*)d_in[8];          // [N]
    const int*   src = ei;
    const int*   dst = ei + N_EDGES;
    float* out = (float*)d_out;

    // Workspace layout
    float* q      = (float*)d_ws;                    // N*64
    float* k      = q + (size_t)N_NODES * H_DIM;     // N*64
    float* v      = k + (size_t)N_NODES * H_DIM;     // N*64
    float* expbuf = v + (size_t)N_NODES * H_DIM;     // E
    float* denom  = expbuf + N_EDGES;                // G
    _Float16* Wt  = (_Float16*)(denom + N_GRAPHS);   // 3*64*256 f16
    _Float16* xh  = Wt + 3 * H_DIM * F_IN;           // N*256 f16

    // 0a) zero accumulators
    {
        int n = N_NODES * H_DIM;
        zero_kernel<<<(n + 255) / 256, 256, 0, stream>>>(out, denom);
    }
    // 0b) weight transpose/convert
    {
        int n = 3 * H_DIM * F_IN;
        wt_kernel<<<(n + 255) / 256, 256, 0, stream>>>(Wq, Wk, Wv, Wt);
    }
    // 0c) x f32 -> f16 (vectorized by 4)
    {
        int n = (N_NODES * F_IN) / 4;
        xh_kernel<<<(n + 255) / 256, 256, 0, stream>>>(x, xh);
    }
    // 1) fused q/k/v projections via WMMA: one wave per 16-node row tile
    {
        proj_gemm_kernel<<<N_NODES / 16, 32, 0, stream>>>(xh, Wt, bq, bk, bv,
                                                          q, k, v);
    }
    // 2) edge scores + per-graph denominators (16 edges per 256-thread block)
    {
        int blocks = (N_EDGES * 16 + 255) / 256;
        score_kernel<<<blocks, 256, 0, stream>>>(q, k, src, dst, bat, expbuf, denom);
    }
    // 3) weighted scatter-add into out
    {
        int blocks = (N_EDGES * 16 + 255) / 256;
        agg_kernel<<<blocks, 256, 0, stream>>>(v, src, dst, bat, expbuf, denom, out);
    }
}
